// Transformer_66477503807603
// MI455X (gfx1250) — matrix operand order repaired and checked by reading, not compiled, not run on hardware
//
#include <hip/hip_runtime.h>
#include <math.h>

// ---------------------------------------------------------------------------
// Model constants (match reference)
// ---------------------------------------------------------------------------
#define DMODEL 512
#define NHEAD  8
#define HDK    64
#define FFN    2048
#define NLAYER 6
#define BATCH  2
#define SEQ    1024
#define VOCAB  32000
#define ROWS   (BATCH * SEQ)     // 2048 activation rows

typedef __bf16 bf16;
typedef __attribute__((ext_vector_type(16))) __bf16 v16bf;
typedef __attribute__((ext_vector_type(8)))  float  v8f;

// CDNA5 async global->LDS copy (GLOBAL_LOAD_ASYNC_TO_LDS_B128, ASYNCcnt).
// lds_off is the raw 32-bit LDS byte offset (flat LDS aperture addr[31:0]).
__device__ __forceinline__ void async_copy_b128(unsigned lds_off, const void* gaddr)
{
    asm volatile("global_load_async_to_lds_b128 %0, %1, off"
                 :: "v"(lds_off), "v"(gaddr) : "memory");
}
__device__ __forceinline__ void wait_asynccnt0()
{
    asm volatile("s_wait_asynccnt 0x0" ::: "memory");
}

// ---------------------------------------------------------------------------
// GEMM: C(MxN) = A(MxK) @ B + [bias] (+ReLU) (+residual), fp32 in/out,
// bf16 WMMA core (v_wmma_f32_16x16x32_bf16), fp32 accumulate.
// flags bit0 = ReLU, bit1 = B is (NxK) row-major (i.e. compute A @ B^T).
// Block: 256 threads = 8 waves (2x4); each wave owns a 32x32 tile (4 WMMAs);
// block tile 64x128, K-step 32.
// ---------------------------------------------------------------------------
#define GBM 64
#define GBN 128
#define GBK 32

__global__ __launch_bounds__(256)
void gemm_bf16_wmma(const float* __restrict__ A, const float* __restrict__ B,
                    float* __restrict__ C, int M, int N, int K,
                    const float* __restrict__ bias,
                    const float* __restrict__ res, int flags)
{
    __shared__ bf16 As[GBM][GBK + 8];   // [m][k], +8 bf16 pad (16B, keeps b128 align)
    __shared__ bf16 Bs[GBN][GBK + 8];   // [n][k] (K-major per output column)

    const int t    = threadIdx.x;
    const int bm   = blockIdx.y * GBM;
    const int bn   = blockIdx.x * GBN;
    const int w    = t >> 5, l = t & 31;
    const int half = l >> 4, lm = l & 15;
    const int wm   = w >> 2, wn = w & 3;           // 2x4 wave grid of 32x32 tiles

    v8f acc[2][2] = {};

    for (int k0 = 0; k0 < K; k0 += GBK) {
        // ---- stage A tile (64x32 fp32 -> bf16) : 8 floats / thread ----
        {
            const int m = t >> 2, kq = (t & 3) * 8;
            const float* ap = A + (size_t)(bm + m) * K + k0 + kq;
            const float4 a0 = *(const float4*)(ap);
            const float4 a1 = *(const float4*)(ap + 4);
            As[m][kq + 0] = (bf16)a0.x; As[m][kq + 1] = (bf16)a0.y;
            As[m][kq + 2] = (bf16)a0.z; As[m][kq + 3] = (bf16)a0.w;
            As[m][kq + 4] = (bf16)a1.x; As[m][kq + 5] = (bf16)a1.y;
            As[m][kq + 6] = (bf16)a1.z; As[m][kq + 7] = (bf16)a1.w;
            if (k0 + GBK < K)
                __builtin_prefetch(ap + GBK, 0, 1);
        }
        // ---- stage B tile (-> Bs[n][k]) : 16 floats / thread ----
        if (flags & 2) {
            // B is (N x K) row-major: K-contiguous reads, direct store
            const int n = t >> 1, kq = (t & 1) * 16;
            const float* bp = B + (size_t)(bn + n) * K + k0 + kq;
            #pragma unroll
            for (int g = 0; g < 4; ++g) {
                const float4 bv = *(const float4*)(bp + g * 4);
                Bs[n][kq + g*4 + 0] = (bf16)bv.x; Bs[n][kq + g*4 + 1] = (bf16)bv.y;
                Bs[n][kq + g*4 + 2] = (bf16)bv.z; Bs[n][kq + g*4 + 3] = (bf16)bv.w;
            }
        } else {
            // B is (K x N) row-major: coalesced row reads, transposed LDS store
            const int kk = t >> 3, nq = (t & 7) * 16;
            const float* bp = B + (size_t)(k0 + kk) * N + bn + nq;
            #pragma unroll
            for (int g = 0; g < 4; ++g) {
                const float4 bv = *(const float4*)(bp + g * 4);
                Bs[nq + g*4 + 0][kk] = (bf16)bv.x; Bs[nq + g*4 + 1][kk] = (bf16)bv.y;
                Bs[nq + g*4 + 2][kk] = (bf16)bv.z; Bs[nq + g*4 + 3][kk] = (bf16)bv.w;
            }
            if (k0 + GBK < K)
                __builtin_prefetch(B + (size_t)(k0 + GBK + kk) * N + bn + nq, 0, 1);
        }
        __syncthreads();

        // ---- fragments: 2 A-frags (rows), 2 B-frags (cols), 4 WMMAs ----
        v16bf af[2], bfv[2];
        #pragma unroll
        for (int i = 0; i < 2; ++i) {
            const bf16* p0 = &As[wm * 32 + i * 16 + lm][half * 8];
            const bf16* p1 = &As[wm * 32 + i * 16 + lm][16 + half * 8];
            #pragma unroll
            for (int j = 0; j < 8; ++j) { af[i][j] = p0[j]; af[i][8 + j] = p1[j]; }
        }
        #pragma unroll
        for (int i = 0; i < 2; ++i) {
            const bf16* p = &Bs[wn * 32 + i * 16 + lm][half * 8];
            #pragma unroll
            for (int j = 0; j < 8; ++j) { bfv[i][j] = p[j]; bfv[i][8 + j] = p[16 + j]; }
        }
        #pragma unroll
        for (int i = 0; i < 2; ++i)
            #pragma unroll
            for (int j = 0; j < 2; ++j)
                acc[i][j] = __builtin_amdgcn_wmma_f32_16x16x32_bf16(
                    false, af[i], false, bfv[j], (short)0, acc[i][j], false, false);
        __syncthreads();
    }

    // ---- epilogue: C rows r+8*half, col lm (ISA C/D layout) ----
    #pragma unroll
    for (int i = 0; i < 2; ++i) {
        const int row0 = bm + wm * 32 + i * 16 + 8 * half;
        #pragma unroll
        for (int j = 0; j < 2; ++j) {
            const int col = bn + wn * 32 + j * 16 + lm;
            const float bb = bias ? bias[col] : 0.0f;
            #pragma unroll
            for (int r = 0; r < 8; ++r) {
                float v = acc[i][j][r] + bb;
                if (flags & 1) v = fmaxf(v, 0.0f);
                if (res) v += res[(size_t)(row0 + r) * N + col];
                C[(size_t)(row0 + r) * N + col] = v;
            }
        }
    }
}

// ---------------------------------------------------------------------------
// Attention: one wave handles 16 query rows of one (batch, head).
// Scores (16 x lkv) in LDS fp32; QK^T and PV via WMMA; softmax fp32 VALU.
// V tiles staged global->LDS with CDNA5 async-DMA copies (ASYNCcnt).
// Q,K,V,O layout: (B, len, NHEAD*HDK) row-major, head offset h*HDK.
// ---------------------------------------------------------------------------
#define VSTRIDE 68   // 64 floats + 16B pad: keeps async b128 LDS writes aligned

__global__ __launch_bounds__(32)
void attn_wmma(const float* __restrict__ Q, const float* __restrict__ K,
               const float* __restrict__ V, float* __restrict__ O,
               int lq, int lkv, int causal)
{
    __shared__ float sc[16][SEQ];          // 64 KB score buffer
    __shared__ float Vs[32][VSTRIDE];      // 8.5 KB staged V chunk (32 keys x 64)

    const int l = threadIdx.x, half = l >> 4, lm = l & 15;
    const int qt = blockIdx.x;             // query tile (16 rows)
    const int bh = blockIdx.y;
    const int b  = bh / NHEAD, h = bh % NHEAD;

    // ---- Q fragments (A-matrix 16x64 -> two 16x32 frags), global fp32 -> bf16 ----
    const size_t qbase = ((size_t)b * lq + qt * 16 + lm) * DMODEL + h * HDK;
    v16bf aq[2];
    #pragma unroll
    for (int f = 0; f < 2; ++f) {
        const float* p0 = Q + qbase + f * 32 + half * 8;
        const float* p1 = Q + qbase + f * 32 + 16 + half * 8;
        #pragma unroll
        for (int j = 0; j < 8; ++j) { aq[f][j] = (bf16)p0[j]; aq[f][8 + j] = (bf16)p1[j]; }
    }

    const int kmax = causal ? (qt * 16 + 16) : lkv;

    // ---- pass 1: scores = Q @ K^T / sqrt(64) ----
    for (int kt = 0; kt * 16 < kmax; ++kt) {
        const float* kr = K + ((size_t)b * lkv + kt * 16 + lm) * DMODEL + h * HDK;
        v8f s = {};
        #pragma unroll
        for (int f = 0; f < 2; ++f) {
            v16bf bk;                                   // B frag: two 8-feat runs (table order)
            const float* p = kr + f * 32 + half * 8;
            #pragma unroll
            for (int j = 0; j < 8; ++j) { bk[j] = (bf16)p[j]; bk[8 + j] = (bf16)p[16 + j]; }
            s = __builtin_amdgcn_wmma_f32_16x16x32_bf16(false, aq[f], false, bk,
                                                        (short)0, s, false, false);
        }
        const int kidx = kt * 16 + lm;
        #pragma unroll
        for (int r = 0; r < 8; ++r) {
            const int m = r + 8 * half;
            float v = s[r] * 0.125f;                    // 1/sqrt(HDK)
            if (causal && kidx > qt * 16 + m) v = -__builtin_inff();
            sc[m][kidx] = v;
        }
    }
    __syncthreads();

    // ---- pass 2: softmax per row (2 lanes / row, interleaved segments) ----
    {
        const int m = lm;
        float mx = -__builtin_inff();
        for (int k = half; k < kmax; k += 2) mx = fmaxf(mx, sc[m][k]);
        mx = fmaxf(mx, __shfl_xor(mx, 16, 32));
        float sum = 0.0f;
        for (int k = half; k < kmax; k += 2) {
            float e = __expf(sc[m][k] - mx);
            sc[m][k] = e; sum += e;
        }
        sum += __shfl_xor(sum, 16, 32);
        const float inv = 1.0f / sum;
        for (int k = half; k < kmax; k += 2) sc[m][k] *= inv;
    }
    __syncthreads();

    // ---- pass 3: O = P @ V (P from LDS; V async-staged global->LDS) ----
    v8f o[4] = {};
    const unsigned vs_base = (unsigned)(size_t)(&Vs[0][0]);   // LDS aperture low 32b
    for (int kc = 0; kc < kmax; kc += 32) {
        // stage V[kc + l][0..15] into Vs[l][*]: 16B per lane per issue, 4 issues
        {
            const int kk = kc + l;
            const int kcl = (kk < lkv) ? kk : (lkv - 1);        // clamp (P==0 there)
            const float* vrow = V + ((size_t)b * lkv + kcl) * DMODEL + h * HDK;
            const unsigned ldst = vs_base + (unsigned)l * (VSTRIDE * 4);
            #pragma unroll
            for (int s = 0; s < 4; ++s)
                async_copy_b128(ldst + s * 16, vrow + s * 4);
        }
        // P A-fragment while the DMA is in flight
        v16bf ap;
        #pragma unroll
        for (int j = 0; j < 8; ++j) {
            const int ka = kc + half * 8 + j;
            const int kb = kc + 16 + half * 8 + j;
            ap[j]     = (ka < kmax) ? (bf16)sc[lm][ka] : (bf16)0.0f;
            ap[8 + j] = (kb < kmax) ? (bf16)sc[lm][kb] : (bf16)0.0f;
        }
        wait_asynccnt0();
        __builtin_amdgcn_s_barrier_signal(-1);
        __builtin_amdgcn_s_barrier_wait(-1);
        #pragma unroll
        for (int nt = 0; nt < 4; ++nt) {                // HDK = 4 x 16 cols
            v16bf bv;                                   // B frag from staged LDS
            #pragma unroll
            for (int j = 0; j < 16; ++j)
                bv[j] = (bf16)Vs[(j < 8 ? half * 8 + j : 16 + half * 8 + (j - 8))][nt * 16 + lm];
            o[nt] = __builtin_amdgcn_wmma_f32_16x16x32_bf16(false, ap, false, bv,
                                                            (short)0, o[nt], false, false);
        }
        __syncthreads();
    }
    #pragma unroll
    for (int nt = 0; nt < 4; ++nt)
        #pragma unroll
        for (int r = 0; r < 8; ++r)
            O[((size_t)b * lq + qt * 16 + r + 8 * half) * DMODEL + h * HDK + nt * 16 + lm]
                = o[nt][r];
}

// ---------------------------------------------------------------------------
// LayerNorm over D=512 per row. Block=256, each thread owns 2 elements.
// ---------------------------------------------------------------------------
__global__ __launch_bounds__(256)
void ln_k(const float* __restrict__ X, float* __restrict__ Y,
          const float* __restrict__ g, const float* __restrict__ be)
{
    __shared__ float sred[8];
    const int row = blockIdx.x, t = threadIdx.x;
    const int wid = t >> 5, lane = t & 31;

    const float x0 = X[(size_t)row * DMODEL + t];
    const float x1 = X[(size_t)row * DMODEL + t + 256];

    float s = x0 + x1;
    #pragma unroll
    for (int off = 16; off; off >>= 1) s += __shfl_xor(s, off, 32);
    if (lane == 0) sred[wid] = s;
    __syncthreads();
    if (t == 0) { float tot = 0; for (int i = 0; i < 8; ++i) tot += sred[i];
                  sred[0] = tot * (1.0f / DMODEL); }
    __syncthreads();
    const float mean = sred[0];
    __syncthreads();

    const float d0 = x0 - mean, d1 = x1 - mean;
    float v = d0 * d0 + d1 * d1;
    #pragma unroll
    for (int off = 16; off; off >>= 1) v += __shfl_xor(v, off, 32);
    if (lane == 0) sred[wid] = v;
    __syncthreads();
    if (t == 0) { float tot = 0; for (int i = 0; i < 8; ++i) tot += sred[i];
                  sred[0] = tot * (1.0f / DMODEL); }
    __syncthreads();
    const float rstd = rsqrtf(sred[0] + 1e-5f);

    Y[(size_t)row * DMODEL + t]       = d0 * rstd * g[t]       + be[t];
    Y[(size_t)row * DMODEL + t + 256] = d1 * rstd * g[t + 256] + be[t + 256];
}

// ---------------------------------------------------------------------------
// RoPE over (B, S, NHEAD*HDK) in fp32. rot_half: [-x2, x1] per 64-wide head.
// ---------------------------------------------------------------------------
__global__ __launch_bounds__(256)
void rope_k(const float* __restrict__ X, float* __restrict__ Y, int total)
{
    int idx = blockIdx.x * 256 + threadIdx.x;
    if (idx >= total) return;
    const int row = idx / DMODEL, d = idx % DMODEL;
    const int pos = row % SEQ;
    const int dk  = d % HDK, i = dk & 31;
    const float freq = __powf(10000.0f, -((float)(2 * i)) * (1.0f / HDK));
    const float ang  = (float)pos * freq;
    const float c = __cosf(ang), sn = __sinf(ang);
    const float x  = X[idx];
    const float xr = (dk < 32) ? -X[idx + 32] : X[idx - 32];
    Y[idx] = x * c + xr * sn;
}

// ---------------------------------------------------------------------------
// Embedding gather + sqrt(D) scale.
// ---------------------------------------------------------------------------
__global__ __launch_bounds__(256)
void embed_k(const int* __restrict__ tok, const float* __restrict__ E,
             float* __restrict__ Y, int total, float scale)
{
    int idx = blockIdx.x * 256 + threadIdx.x;
    if (idx >= total) return;
    const int row = idx / DMODEL, d = idx % DMODEL;
    Y[idx] = E[(size_t)tok[row] * DMODEL + d] * scale;
}

// ---------------------------------------------------------------------------
// Orchestration
// ---------------------------------------------------------------------------
extern "C" void kernel_launch(void* const* d_in, const int* in_sizes, int n_in,
                              void* d_out, int out_size, void* d_ws, size_t ws_size,
                              hipStream_t stream)
{
    const float* embed   = (const float*)d_in[0];
    const float* eattn   = (const float*)d_in[1];
    const float* ew1     = (const float*)d_in[2];
    const float* eb1     = (const float*)d_in[3];
    const float* ew2     = (const float*)d_in[4];
    const float* eb2     = (const float*)d_in[5];
    const float* elng    = (const float*)d_in[6];
    const float* elnb    = (const float*)d_in[7];
    const float* efg     = (const float*)d_in[8];
    const float* efb     = (const float*)d_in[9];
    const float* dself   = (const float*)d_in[10];
    const float* dcross  = (const float*)d_in[11];
    const float* dw1     = (const float*)d_in[12];
    const float* db1     = (const float*)d_in[13];
    const float* dw2     = (const float*)d_in[14];
    const float* db2     = (const float*)d_in[15];
    const float* dlng    = (const float*)d_in[16];
    const float* dlnb    = (const float*)d_in[17];
    const float* dfg     = (const float*)d_in[18];
    const float* dfb     = (const float*)d_in[19];
    const int*   src     = (const int*)d_in[20];
    const int*   tgt     = (const int*)d_in[21];
    float*       out     = (float*)d_out;

    const size_t R  = (size_t)ROWS * DMODEL;   // 1M floats per activation tensor
    float* ws  = (float*)d_ws;
    float* x   = ws + 0 * R;
    float* n   = ws + 1 * R;
    float* q   = ws + 2 * R;
    float* k   = ws + 3 * R;
    float* v   = ws + 4 * R;
    float* qr  = ws + 5 * R;
    float* kr  = ws + 6 * R;
    float* ctx = ws + 7 * R;
    float* enc = ws + 8 * R;
    float* y   = ws + 9 * R;
    float* hb  = ws + 10 * R;                  // FFN hidden: 4R floats

    const int   DD    = DMODEL * DMODEL;
    const float scale = sqrtf((float)DMODEL);
    const int   tot   = (int)R;

    auto gemm = [&](const float* A, const float* B, float* C, int M, int N, int K,
                    const float* bias, const float* res, int flags) {
        dim3 grid(N / GBN, M / GBM);
        gemm_bf16_wmma<<<grid, 256, 0, stream>>>(A, B, C, M, N, K, bias, res, flags);
    };
    auto ln = [&](const float* X, float* Y, const float* g, const float* b) {
        ln_k<<<ROWS, 256, 0, stream>>>(X, Y, g, b);
    };
    auto rope = [&](const float* X, float* Y) {
        rope_k<<<(tot + 255) / 256, 256, 0, stream>>>(X, Y, tot);
    };
    auto attn = [&](const float* Qp, const float* Kp, const float* Vp, float* Op,
                    int causal) {
        dim3 grid(SEQ / 16, BATCH * NHEAD);
        attn_wmma<<<grid, 32, 0, stream>>>(Qp, Kp, Vp, Op, SEQ, SEQ, causal);
    };

    // ================= encoder =================
    embed_k<<<(tot + 255) / 256, 256, 0, stream>>>(src, embed, x, tot, scale);
    for (int i = 0; i < NLAYER; ++i) {
        const float* W = eattn + (size_t)i * 4 * DD;
        ln(x, n, elng + (size_t)(i * 2 + 0) * DMODEL, elnb + (size_t)(i * 2 + 0) * DMODEL);
        gemm(n, W + 0 * DD, q, ROWS, DMODEL, DMODEL, nullptr, nullptr, 0);
        gemm(n, W + 1 * DD, k, ROWS, DMODEL, DMODEL, nullptr, nullptr, 0);
        gemm(n, W + 2 * DD, v, ROWS, DMODEL, DMODEL, nullptr, nullptr, 0);
        rope(q, qr); rope(k, kr);
        attn(qr, kr, v, ctx, 0);
        gemm(ctx, W + 3 * DD, x, ROWS, DMODEL, DMODEL, nullptr, x, 0);
        ln(x, n, elng + (size_t)(i * 2 + 1) * DMODEL, elnb + (size_t)(i * 2 + 1) * DMODEL);
        gemm(n, ew1 + (size_t)i * DMODEL * FFN, hb, ROWS, FFN, DMODEL,
             eb1 + (size_t)i * FFN, nullptr, 1 /*relu*/);
        gemm(hb, ew2 + (size_t)i * FFN * DMODEL, x, ROWS, DMODEL, FFN,
             eb2 + (size_t)i * DMODEL, x, 0);
    }
    ln(x, enc, efg, efb);

    // ================= decoder =================
    embed_k<<<(tot + 255) / 256, 256, 0, stream>>>(tgt, embed, y, tot, scale);
    for (int i = 0; i < NLAYER; ++i) {
        const float* Ws = dself  + (size_t)i * 4 * DD;
        const float* Wc = dcross + (size_t)i * 4 * DD;
        // self-attention (RoPE + causal)
        ln(y, n, dlng + (size_t)(i * 3 + 0) * DMODEL, dlnb + (size_t)(i * 3 + 0) * DMODEL);
        gemm(n, Ws + 0 * DD, q, ROWS, DMODEL, DMODEL, nullptr, nullptr, 0);
        gemm(n, Ws + 1 * DD, k, ROWS, DMODEL, DMODEL, nullptr, nullptr, 0);
        gemm(n, Ws + 2 * DD, v, ROWS, DMODEL, DMODEL, nullptr, nullptr, 0);
        rope(q, qr); rope(k, kr);
        attn(qr, kr, v, ctx, 1);
        gemm(ctx, Ws + 3 * DD, y, ROWS, DMODEL, DMODEL, nullptr, y, 0);
        // cross-attention (no RoPE, no mask; K/V from encoder output)
        ln(y, n, dlng + (size_t)(i * 3 + 1) * DMODEL, dlnb + (size_t)(i * 3 + 1) * DMODEL);
        gemm(n,   Wc + 0 * DD, q, ROWS, DMODEL, DMODEL, nullptr, nullptr, 0);
        gemm(enc, Wc + 1 * DD, k, ROWS, DMODEL, DMODEL, nullptr, nullptr, 0);
        gemm(enc, Wc + 2 * DD, v, ROWS, DMODEL, DMODEL, nullptr, nullptr, 0);
        attn(q, k, v, ctx, 0);
        gemm(ctx, Wc + 3 * DD, y, ROWS, DMODEL, DMODEL, nullptr, y, 0);
        // FFN
        ln(y, n, dlng + (size_t)(i * 3 + 2) * DMODEL, dlnb + (size_t)(i * 3 + 2) * DMODEL);
        gemm(n, dw1 + (size_t)i * DMODEL * FFN, hb, ROWS, FFN, DMODEL,
             db1 + (size_t)i * FFN, nullptr, 1 /*relu*/);
        gemm(hb, dw2 + (size_t)i * FFN * DMODEL, y, ROWS, DMODEL, FFN,
             db2 + (size_t)i * DMODEL, y, 0);
    }
    ln(y, n, dfg, dfb);

    // ====== output projection: logits = y_ln @ embed^T  (B-transposed GEMM) ======
    gemm(n, embed, out, ROWS, VOCAB, DMODEL, nullptr, nullptr, 2 /*B^T*/);
}
